// SpeakerDecoder_57054345560170
// MI455X (gfx1250) — matrix operand order repaired and checked by reading, not compiled, not run on hardware
//
#include <hip/hip_runtime.h>
#include <hip/hip_bf16.h>
#include <math.h>

#define HIDDEN   512
#define BATCH    4096
#define MSG_LEN  16
#define ALPHA    64
#define PAD_TOK  64
#define BOS_TOK  65
#define EOS_TOK  0
#define G4       2048   // 4*HIDDEN

typedef __bf16 v16bf __attribute__((ext_vector_type(16)));
typedef __bf16 v8bf  __attribute__((ext_vector_type(8)));
typedef float  v8f   __attribute__((ext_vector_type(8)));

#define LDA 40   // padded LDS row stride (bf16 elems): 80B -> breaks bank-conflict cycle

__device__ __forceinline__ unsigned short f2bf(float f) {
    unsigned int u = __float_as_uint(f);
    unsigned int r = u + 0x7FFFu + ((u >> 16) & 1u);   // round-to-nearest-even
    return (unsigned short)(r >> 16);
}
__device__ __forceinline__ float sigmoidf(float x) { return 1.0f / (1.0f + __expf(-x)); }

__device__ __forceinline__ unsigned int hash3(unsigned int t, unsigned int b, unsigned int a) {
    unsigned int x = t * 0x85EBCA6Bu ^ b * 0xC2B2AE35u ^ a * 0x27D4EB2Fu ^ 0x9E3779B9u;
    x ^= x >> 16; x *= 0x7FEB352Du;
    x ^= x >> 15; x *= 0x846CA68Bu;
    x ^= x >> 16;
    return x;
}

// ---------------- conversion / setup kernels ----------------

__global__ void k_cvt_bf16(const float* __restrict__ src, unsigned short* __restrict__ dst, int n) {
    int i = blockIdx.x * blockDim.x + threadIdx.x;
    if (i < n) dst[i] = f2bf(src[i]);
}

__global__ void k_zero_row(unsigned short* __restrict__ dst) {
    dst[blockIdx.x * blockDim.x + threadIdx.x] = 0;   // 2 blocks x 256 = 512 elems
}

__global__ void k_bias(const float* __restrict__ a, const float* __restrict__ b,
                       float* __restrict__ out) {
    int i = blockIdx.x * blockDim.x + threadIdx.x;
    if (i < G4) out[i] = a[i] + b[i];
}

__global__ void k_init(const float* __restrict__ encoded, const unsigned short* __restrict__ embb,
                       float* __restrict__ c, unsigned short* __restrict__ hb,
                       unsigned short* __restrict__ xb, int* __restrict__ stopped) {
    int i = blockIdx.x * blockDim.x + threadIdx.x;    // over BATCH*HIDDEN
    int j = i & (HIDDEN - 1);
    float e = encoded[i];
    c[i]  = e;
    hb[i] = f2bf(e);
    xb[i] = embb[BOS_TOK * HIDDEN + j];
    if (j == 0) stopped[i >> 9] = 0;
}

// ---------------- WMMA GEMM: gates = X@Wih^T + H@Whh^T + bias ----------------
// Workgroup (8 waves) computes a 128(M) x 64(N) block staged through LDS.
// Wave grid 4(M) x 2(N); each wave holds a 32x32 register tile (4 x v8f).

__global__ void __launch_bounds__(256)
k_gemm_gates(const unsigned short* __restrict__ X,   const unsigned short* __restrict__ H,
             const unsigned short* __restrict__ Wih, const unsigned short* __restrict__ Whh,
             const float* __restrict__ bias, float* __restrict__ gates) {
    __shared__ unsigned short lA[128 * LDA];   // 128 rows (M) x 32 K, padded
    __shared__ unsigned short lB[64 * LDA];    // 64 rows (N)  x 32 K, padded

    const int tid  = threadIdx.x;
    const int lane = tid & 31;
    const int wv   = tid >> 5;          // 0..7
    const int lh   = lane >> 4;         // 0/1
    const int l15  = lane & 15;
    const int wm   = wv & 3;            // wave M index: 0..3 (32 rows each)
    const int wn   = wv >> 2;           // wave N index: 0..1 (32 cols each)
    const int bm   = (blockIdx.x & 31) * 128;   // 4096/128 = 32 M-blocks
    const int bn   = (blockIdx.x >> 5) * 64;    // 2048/64  = 32 N-blocks

    v8f acc[2][2] = {};

    // staging index split (constant across phases)
    const int row_a = tid >> 1, ha = tid & 1;   // 128 rows x 2 halves of 16 elems
    const int row_b = tid >> 2, qb = tid & 3;   // 64 rows x 4 quarters of 8 elems
    unsigned short* sA = &lA[row_a * LDA + ha * 16];
    unsigned short* sB = &lB[row_b * LDA + qb * 8];

    auto phase = [&](const unsigned short* __restrict__ Ain,
                     const unsigned short* __restrict__ Win) {
        const unsigned short* gA = Ain + (size_t)(bm + row_a) * HIDDEN + ha * 16;
        const unsigned short* gB = Win + (size_t)(bn + row_b) * HIDDEN + qb * 8;
        #pragma unroll 2
        for (int kb = 0; kb < HIDDEN; kb += 32) {
            __builtin_prefetch(gA + kb + 32, 0, 1);
            __builtin_prefetch(gB + kb + 32, 0, 1);
            v8bf a0 = *reinterpret_cast<const v8bf*>(gA + kb);
            v8bf a1 = *reinterpret_cast<const v8bf*>(gA + kb + 8);
            v8bf b0 = *reinterpret_cast<const v8bf*>(gB + kb);
            *reinterpret_cast<v8bf*>(sA)     = a0;
            *reinterpret_cast<v8bf*>(sA + 8) = a1;
            *reinterpret_cast<v8bf*>(sB)     = b0;
            __syncthreads();

            v16bf afrag[2], bfrag[2];
            #pragma unroll
            for (int ms = 0; ms < 2; ++ms) {
                const unsigned short* p = &lA[(wm * 32 + ms * 16 + l15) * LDA + lh * 8];
                v8bf lo = *reinterpret_cast<const v8bf*>(p);        // K 0-7  / 8-15
                v8bf hi = *reinterpret_cast<const v8bf*>(p + 16);   // K 16-23 / 24-31
                #pragma unroll
                for (int i = 0; i < 8; ++i) { afrag[ms][i] = lo[i]; afrag[ms][8 + i] = hi[i]; }
            }
            #pragma unroll
            for (int ns = 0; ns < 2; ++ns) {
                const unsigned short* p = &lB[(wn * 32 + ns * 16 + l15) * LDA + lh * 8];
                v8bf lo = *reinterpret_cast<const v8bf*>(p);        // K 0-7  / 8-15
                v8bf hi = *reinterpret_cast<const v8bf*>(p + 16);   // K 16-23 / 24-31
                #pragma unroll
                for (int i = 0; i < 8; ++i) { bfrag[ns][i] = lo[i]; bfrag[ns][8 + i] = hi[i]; }
            }
            #pragma unroll
            for (int ms = 0; ms < 2; ++ms)
                #pragma unroll
                for (int ns = 0; ns < 2; ++ns)
                    acc[ms][ns] = __builtin_amdgcn_wmma_f32_16x16x32_bf16(
                        false, afrag[ms], false, bfrag[ns], (short)0, acc[ms][ns], false, false);
            __syncthreads();
        }
    };

    phase(X, Wih);
    phase(H, Whh);

    #pragma unroll
    for (int ms = 0; ms < 2; ++ms) {
        #pragma unroll
        for (int ns = 0; ns < 2; ++ns) {
            const int col = bn + wn * 32 + ns * 16 + l15;
            const float bcol = bias[col];
            #pragma unroll
            for (int r = 0; r < 8; ++r) {
                const int row = bm + wm * 32 + ms * 16 + r + lh * 8;
                gates[(size_t)row * G4 + col] = acc[ms][ns][r] + bcol;
            }
        }
    }
}

// ---------------- LSTM cell elementwise ----------------

__global__ void k_cell(const float* __restrict__ gates, float* __restrict__ c,
                       unsigned short* __restrict__ hb) {
    int i = blockIdx.x * blockDim.x + threadIdx.x;    // over BATCH*HIDDEN
    int b = i >> 9, j = i & 511;
    const float* gr = gates + (size_t)b * G4;
    float gi = gr[j], gf = gr[512 + j], gg = gr[1024 + j], go = gr[1536 + j];
    float cn = sigmoidf(gf) * c[i] + sigmoidf(gi) * tanhf(gg);
    float hn = sigmoidf(go) * tanhf(cn);
    c[i]  = cn;
    hb[i] = f2bf(hn);
}

// ---------------- WMMA GEMM: logits = H@Wp^T + bp (N=64, direct) ----------------

__global__ void __launch_bounds__(256)
k_gemm_logits(const unsigned short* __restrict__ H, const unsigned short* __restrict__ Wp,
              const float* __restrict__ bp, float* __restrict__ logits) {
    const int wid  = blockIdx.x * 8 + (threadIdx.x >> 5);
    const int lane = threadIdx.x & 31;
    const int lh   = lane >> 4;
    const int l15  = lane & 15;
    const int mt   = wid >> 2;         // 64/16 = 4 N-tiles
    const int nt   = wid & 3;
    const int m0   = mt * 16, n0 = nt * 16;

    v8f acc = {};
    const unsigned short* arow = H  + (size_t)(m0 + l15) * HIDDEN + lh * 16;
    const unsigned short* brow = Wp + (size_t)(n0 + l15) * HIDDEN + lh * 16;
    #pragma unroll
    for (int kb = 0; kb < HIDDEN; kb += 32) {
        v8bf alo = *reinterpret_cast<const v8bf*>(arow + kb);
        v8bf ahi = *reinterpret_cast<const v8bf*>(arow + kb + 8);
        v16bf a;
        #pragma unroll
        for (int i = 0; i < 8; ++i) { a[i] = alo[i]; a[8 + i] = ahi[i]; }
        v16bf b = *reinterpret_cast<const v16bf*>(brow + kb);
        acc = __builtin_amdgcn_wmma_f32_16x16x32_bf16(false, a, false, b,
                                                      (short)0, acc, false, false);
    }
    const float bcol = bp[n0 + l15];
    #pragma unroll
    for (int r = 0; r < 8; ++r) {
        logits[(size_t)(m0 + r + lh * 8) * ALPHA + n0 + l15] = acc[r] + bcol;
    }
}

// ---------------- softmax / Gumbel-max sampling ----------------

__global__ void k_sample(const float* __restrict__ logits, int* __restrict__ stopped,
                         int* __restrict__ action, int* __restrict__ msg,
                         float* __restrict__ lp, float* __restrict__ entbuf, int t) {
    int b = blockIdx.x * blockDim.x + threadIdx.x;    // over BATCH
    const float* row = logits + (size_t)b * ALPHA;

    float m = -3.402823466e38f;
    for (int a = 0; a < ALPHA; ++a) m = fmaxf(m, row[a]);

    float s = 0.f, dot = 0.f, best = -3.402823466e38f, bl = 0.f;
    int act = 0;
    for (int a = 0; a < ALPHA; ++a) {
        float l = row[a];
        float e = __expf(l - m);
        s   += e;
        dot += e * l;
        float u = (float)(hash3((unsigned)t, (unsigned)b, (unsigned)a) >> 8)
                    * (1.0f / 16777216.0f) + 0.5f / 16777216.0f;
        float g = -__logf(-__logf(u));                // Gumbel(0,1)
        float v = l + g;
        if (v > best) { best = v; act = a; bl = l; }
    }
    float lse  = m + __logf(s);
    int   st   = stopped[b];
    float live = st ? 0.f : 1.f;
    float lpv  = (bl - lse) * live;
    float entv = (lse - dot / s) * live;
    int a2 = st ? PAD_TOK : act;
    stopped[b] = st | (a2 == EOS_TOK);
    action[b]  = a2;
    msg   [b * MSG_LEN + t] = a2;
    lp    [b * MSG_LEN + t] = lpv;
    entbuf[b * MSG_LEN + t] = entv;
}

// ---------------- embedding gather ----------------

__global__ void k_gather(const int* __restrict__ action, const unsigned short* __restrict__ embb,
                         unsigned short* __restrict__ xb) {
    int i = blockIdx.x * blockDim.x + threadIdx.x;    // over BATCH*HIDDEN
    int b = i >> 9, j = i & 511;
    xb[i] = embb[action[b] * HIDDEN + j];
}

// ---------------- finalize ----------------

__global__ void k_final(const int* __restrict__ msg, const float* __restrict__ entbuf,
                        int* __restrict__ msg_len, float* __restrict__ ent_avg) {
    int b = blockIdx.x * blockDim.x + threadIdx.x;    // over BATCH
    int len = 0; float es = 0.f;
    for (int t = 0; t < MSG_LEN; ++t) {
        len += (msg[b * MSG_LEN + t] != PAD_TOK);
        es  += entbuf[b * MSG_LEN + t];
    }
    msg_len[b] = len;
    ent_avg[b] = es / (float)len;
}

// ---------------- host side ----------------

extern "C" void kernel_launch(void* const* d_in, const int* in_sizes, int n_in,
                              void* d_out, int out_size, void* d_ws, size_t ws_size,
                              hipStream_t stream) {
    (void)in_sizes; (void)n_in; (void)out_size; (void)ws_size;

    const float* encoded = (const float*)d_in[0];
    const float* emb     = (const float*)d_in[1];
    const float* Wih     = (const float*)d_in[2];
    const float* Whh     = (const float*)d_in[3];
    const float* bih     = (const float*)d_in[4];
    const float* bhh     = (const float*)d_in[5];
    const float* Wp      = (const float*)d_in[6];
    const float* bp      = (const float*)d_in[7];

    char* ws = (char*)d_ws;
    size_t off = 0;
    auto alloc = [&](size_t bytes) -> void* {
        void* p = ws + off;
        off += (bytes + 255) & ~(size_t)255;
        return p;
    };

    unsigned short* WihB = (unsigned short*)alloc((size_t)G4 * HIDDEN * 2);
    unsigned short* WhhB = (unsigned short*)alloc((size_t)G4 * HIDDEN * 2);
    unsigned short* WpB  = (unsigned short*)alloc((size_t)ALPHA * HIDDEN * 2);
    unsigned short* EmbB = (unsigned short*)alloc((size_t)66 * HIDDEN * 2);
    float*          Bias = (float*)alloc((size_t)G4 * 4);
    unsigned short* Xb   = (unsigned short*)alloc((size_t)BATCH * HIDDEN * 2);
    unsigned short* Hb   = (unsigned short*)alloc((size_t)BATCH * HIDDEN * 2);
    float*          C    = (float*)alloc((size_t)BATCH * HIDDEN * 4);
    float*          Gates= (float*)alloc((size_t)BATCH * G4 * 4);
    float*          Log  = (float*)alloc((size_t)BATCH * ALPHA * 4);
    int*            Act  = (int*)alloc((size_t)BATCH * 4);
    int*            Stop = (int*)alloc((size_t)BATCH * 4);
    float*          Ent  = (float*)alloc((size_t)BATCH * MSG_LEN * 4);

    int*   msg_o = (int*)d_out;
    int*   len_o = (int*)((char*)d_out + (size_t)BATCH * MSG_LEN * 4);
    float* lp_o  = (float*)((char*)d_out + (size_t)(BATCH * MSG_LEN + BATCH) * 4);
    float* ea_o  = (float*)((char*)d_out + (size_t)(2 * BATCH * MSG_LEN + BATCH) * 4);

    // weight / embedding conversion to bf16 (recomputed every call: deterministic)
    k_cvt_bf16<<<(G4 * HIDDEN + 255) / 256, 256, 0, stream>>>(Wih, WihB, G4 * HIDDEN);
    k_cvt_bf16<<<(G4 * HIDDEN + 255) / 256, 256, 0, stream>>>(Whh, WhhB, G4 * HIDDEN);
    k_cvt_bf16<<<(ALPHA * HIDDEN + 255) / 256, 256, 0, stream>>>(Wp, WpB, ALPHA * HIDDEN);
    k_cvt_bf16<<<(66 * HIDDEN + 255) / 256, 256, 0, stream>>>(emb, EmbB, 66 * HIDDEN);
    k_zero_row<<<2, 256, 0, stream>>>(EmbB + (size_t)PAD_TOK * HIDDEN);
    k_bias<<<(G4 + 255) / 256, 256, 0, stream>>>(bih, bhh, Bias);
    k_init<<<BATCH * HIDDEN / 256, 256, 0, stream>>>(encoded, EmbB, C, Hb, Xb, Stop);

    for (int t = 0; t < MSG_LEN; ++t) {
        // 32 x 32 blocks of 128x64 -> 1024 workgroups
        k_gemm_gates<<<(BATCH / 128) * (G4 / 64), 256, 0, stream>>>(Xb, Hb, WihB, WhhB, Bias, Gates);
        k_cell<<<BATCH * HIDDEN / 256, 256, 0, stream>>>(Gates, C, Hb);
        k_gemm_logits<<<(BATCH / 16) * (ALPHA / 16) / 8, 256, 0, stream>>>(Hb, WpB, bp, Log);
        k_sample<<<BATCH / 256, 256, 0, stream>>>(Log, Stop, Act, msg_o, lp_o, Ent, t);
        k_gather<<<BATCH * HIDDEN / 256, 256, 0, stream>>>(Act, EmbB, Xb);
    }
    k_final<<<BATCH / 256, 256, 0, stream>>>(msg_o, Ent, len_o, ea_o);
}
